// CrossAttention_58703613002115
// MI455X (gfx1250) — matrix operand order repaired and checked by reading, not compiled, not run on hardware
//
#include <hip/hip_runtime.h>
#include <hip/hip_bf16.h>

typedef __bf16 bf16;
typedef bf16  v16bf __attribute__((ext_vector_type(16)));
typedef float v8f   __attribute__((ext_vector_type(8)));

#define DIM     2048
#define CTXDIM  1024
#define NHEADS  32
#define HDIM    64
#define BB      2
#define NN      2048
#define CTXN    2048

// ---- CDNA5 async global->LDS copy (ASYNCcnt-tracked DMA path) ----
__device__ __forceinline__ void async_copy16(const bf16* gsrc, bf16* ldst) {
  unsigned loff = (unsigned)(unsigned long long)ldst;
  asm volatile("global_load_async_to_lds_b128 %0, %1, off"
               :: "v"(loff), "v"(gsrc) : "memory");
}
__device__ __forceinline__ void wait_async0() {
  asm volatile("s_wait_asynccnt 0" ::: "memory");
}

// ---- WMMA bf16 fragment loaders (ISA 7.12.2 layouts) ----
// A 16x32 bf16: lane half h; e<8 -> K = kbase + 8h + e ; e>=8 -> K = kbase + 16 + 8h + (e-8)
__device__ __forceinline__ v16bf load_fragA(const bf16* rowp, int kbase, int h) {
  union { v16bf v; uint4 u[2]; } f;
  f.u[0] = *(const uint4*)(rowp + kbase + h * 8);
  f.u[1] = *(const uint4*)(rowp + kbase + 16 + h * 8);
  return f.v;
}
// B 32x16 bf16: lane = column N; e<8 -> K = kbase + 8h + e ; e>=8 -> K = kbase + 16 + 8h + (e-8)
__device__ __forceinline__ v16bf load_fragB(const bf16* rowp, int kbase, int h) {
  union { v16bf v; uint4 u[2]; } f;
  f.u[0] = *(const uint4*)(rowp + kbase + h * 8);
  f.u[1] = *(const uint4*)(rowp + kbase + 16 + h * 8);
  return f.v;
}

// ---- elementwise converts ----
__global__ void cvt_bf16(const float* __restrict__ s, bf16* __restrict__ d, int n) {
  int i = blockIdx.x * blockDim.x + threadIdx.x;
  if (i < n) d[i] = (bf16)s[i];
}
// Wt[n*K + k] = W[k*N + n]
__global__ void cvt_tr(const float* __restrict__ W, bf16* __restrict__ Wt, int K, int N) {
  int i = blockIdx.x * blockDim.x + threadIdx.x;
  if (i >= N * K) return;
  int n = i / K, k = i - n * K;
  Wt[i] = (bf16)W[(size_t)k * N + n];
}

// ---- bf16 WMMA GEMM: async-LDS double-buffered 64x64 B tile, 8 WMMAs per sync ----
// C(MxN fp32 rm) = A(MxK bf16 rm) x Bt(N-major bf16) + bias. Block: 8 waves = 128x64 C tile.
__global__ void __launch_bounds__(256)
gemm_bf16(const bf16* __restrict__ A, const bf16* __restrict__ Bt,
          float* __restrict__ C, const float* __restrict__ bias,
          int M, int N, int K) {
  __shared__ __align__(16) bf16 Bs[2][64 * 64];   // [n in tile][k in tile]
  int tid  = threadIdx.x;
  int lane = tid & 31, wid = tid >> 5;
  int m = lane & 15, hh = lane >> 4;
  int mbase = blockIdx.y * 128 + wid * 16;
  int nbase = blockIdx.x * 64;
  const bf16* arow = A + (size_t)(mbase + m) * K;
  // staging: 64x64 bf16 tile = 512 x 16B chunks; thread t stages chunks t and t+256
  int sn = tid >> 3, sc = (tid & 7) * 8;
  const bf16* bsrc = Bt + (size_t)(nbase + sn) * K + sc;
  size_t brow32 = (size_t)32 * K;

  v8f acc[4] = {};
  int nk = K >> 6;

  // prologue: stage k-tile 0 into buffer 0
  async_copy16(bsrc,          &Bs[0][sn * 64 + sc]);
  async_copy16(bsrc + brow32, &Bs[0][(sn + 32) * 64 + sc]);
  wait_async0();
  __syncthreads();

  int buf = 0;
  for (int kt = 0; kt < nk; ++kt) {
    if (kt + 1 < nk) {                        // DMA next tile while computing this one
      const bf16* s0 = bsrc + ((size_t)(kt + 1) << 6);
      bf16* d0 = &Bs[buf ^ 1][sn * 64 + sc];
      async_copy16(s0,          d0);
      async_copy16(s0 + brow32, d0 + 32 * 64);
    }
    int k0 = kt << 6;
    v16bf a0 = load_fragA(arow, k0, hh);
    v16bf a1 = load_fragA(arow, k0 + 32, hh);
    v16bf bf0[4], bf1[4];
#pragma unroll
    for (int t = 0; t < 4; ++t) {             // batch all LDS fragment loads
      const bf16* brow = &Bs[buf][(t * 16 + m) * 64];
      bf0[t] = load_fragB(brow, 0, hh);
      bf1[t] = load_fragB(brow, 32, hh);
    }
#pragma unroll
    for (int t = 0; t < 4; ++t) {             // dense WMMA burst
      acc[t] = __builtin_amdgcn_wmma_f32_16x16x32_bf16(false, a0, false, bf0[t],
                                                       (short)0, acc[t], false, false);
      acc[t] = __builtin_amdgcn_wmma_f32_16x16x32_bf16(false, a1, false, bf1[t],
                                                       (short)0, acc[t], false, false);
    }
    wait_async0();
    __syncthreads();
    buf ^= 1;
  }
#pragma unroll
  for (int t = 0; t < 4; ++t) {
    int col = nbase + t * 16 + m;
    float bv = bias ? bias[col] : 0.0f;
#pragma unroll
    for (int r = 0; r < 8; ++r) {
      int row = mbase + r + 8 * hh;
      C[(size_t)row * N + col] = acc[t][r] + bv;
    }
  }
}

// ---- axial RoPE: fp32 (b,n,DIM) -> bf16 head-major (b,h,n,64) ----
__global__ void rope_q(const float* __restrict__ Qf, const float* __restrict__ pos,
                       bf16* __restrict__ Qb) {
  int i = blockIdx.x * blockDim.x + threadIdx.x;
  if (i >= BB * NN * NHEADS * 32) return;
  int j = i & 15, p = (i >> 4) & 1, h = (i >> 5) & 31, n = (i >> 10) & (NN - 1), b = i >> 21;
  float freq = __expf(-(float)j * 0.57564627324851f);   // 10000^(-j/16)
  float ang = pos[(b * NN + n) * 2 + p] * freq;
  float sy, cy; __sincosf(ang, &sy, &cy);
  size_t qi = ((size_t)(b * NN + n)) * DIM + h * HDIM + p * 32 + 2 * j;
  float x1 = Qf[qi], x2 = Qf[qi + 1];
  size_t qo = (((size_t)(b * NHEADS + h)) * NN + n) * HDIM + p * 32 + 2 * j;
  Qb[qo]     = (bf16)(x1 * cy - x2 * sy);
  Qb[qo + 1] = (bf16)(x1 * sy + x2 * cy);
}
__global__ void rope_k(const float* __restrict__ KVf, const float* __restrict__ pos,
                       bf16* __restrict__ Kb) {
  int i = blockIdx.x * blockDim.x + threadIdx.x;
  if (i >= BB * CTXN * NHEADS * 32) return;
  int j = i & 15, p = (i >> 4) & 1, h = (i >> 5) & 31, n = (i >> 10) & (CTXN - 1), b = i >> 21;
  float freq = __expf(-(float)j * 0.57564627324851f);
  float ang = pos[(b * CTXN + n) * 2 + p] * freq;
  float sy, cy; __sincosf(ang, &sy, &cy);
  size_t ki = ((size_t)(b * CTXN + n)) * (2 * DIM) + h * HDIM + p * 32 + 2 * j;
  float x1 = KVf[ki], x2 = KVf[ki + 1];
  size_t ko = (((size_t)(b * NHEADS + h)) * CTXN + n) * HDIM + p * 32 + 2 * j;
  Kb[ko]     = (bf16)(x1 * cy - x2 * sy);
  Kb[ko + 1] = (bf16)(x1 * sy + x2 * cy);
}
// V = cols [2048,4096) of KV; transpose to (b, h*64+d, ctx) bf16
__global__ void vtrans(const float* __restrict__ KVf, bf16* __restrict__ Vt) {
  int i = blockIdx.x * blockDim.x + threadIdx.x;
  if (i >= BB * DIM * CTXN) return;
  int ctx = i & (CTXN - 1), c = (i >> 11) & (DIM - 1), b = i >> 22;
  Vt[i] = (bf16)KVf[((size_t)(b * CTXN + ctx)) * (2 * DIM) + DIM + c];
}

// ---- flash attention: block = (b,h) + 128 q rows; K/V chunks async-staged, shared by 8 waves ----
__global__ void __launch_bounds__(256)
attention(const bf16* __restrict__ Qb, const bf16* __restrict__ Kb,
          const bf16* __restrict__ Vt, bf16* __restrict__ attn) {
  __shared__ __align__(16) bf16 Ks[2][32 * 64];   // [ctx in chunk][hd]
  __shared__ __align__(16) bf16 Vs[2][64 * 32];   // [hd][ctx in chunk]
  __shared__ __align__(16) bf16 Ps[8][16 * 32];   // per-wave P tile
  int tid  = threadIdx.x;
  int lane = tid & 31, wid = tid >> 5;
  int m = lane & 15, hh = lane >> 4;
  int qg = blockIdx.x & 15;
  int h  = (blockIdx.x >> 4) & 31;
  int b  = blockIdx.x >> 9;
  int qbase = qg * 128 + wid * 16;

  const bf16* kb = Kb + ((size_t)(b * NHEADS + h)) * CTXN * HDIM;
  const bf16* vb = Vt + ((size_t)(b * NHEADS + h)) * HDIM * CTXN;
  const bf16* qrow = Qb + (((size_t)(b * NHEADS + h)) * NN + qbase + m) * HDIM;
  v16bf a0 = load_fragA(qrow, 0, hh);
  v16bf a1 = load_fragA(qrow, 32, hh);

  int kr = tid >> 3, kc = (tid & 7) * 8;   // K chunk 32x64: 8 x 16B per row
  int vr = tid >> 2, vc = (tid & 3) * 8;   // V chunk 64x32: 4 x 16B per row

  float mrow[8], lrow[8];
  v8f O[4] = {};
#pragma unroll
  for (int r = 0; r < 8; ++r) { mrow[r] = -1.0e30f; lrow[r] = 0.0f; }
  bf16* P = &Ps[wid][0];

  async_copy16(kb + (size_t)kr * HDIM + kc, &Ks[0][kr * 64 + kc]);
  async_copy16(vb + (size_t)vr * CTXN + vc, &Vs[0][vr * 32 + vc]);
  wait_async0();
  __syncthreads();

  int buf = 0;
  for (int c0 = 0; c0 < CTXN; c0 += 32) {
    if (c0 + 32 < CTXN) {                   // DMA next K/V chunk during compute
      int nb = buf ^ 1;
      async_copy16(kb + (size_t)(c0 + 32 + kr) * HDIM + kc, &Ks[nb][kr * 64 + kc]);
      async_copy16(vb + (size_t)vr * CTXN + c0 + 32 + vc,   &Vs[nb][vr * 32 + vc]);
    }
    // batch K-fragments, then S WMMA burst
    const bf16* k0p = &Ks[buf][m * 64];
    const bf16* k1p = &Ks[buf][(16 + m) * 64];
    v16bf kf0 = load_fragB(k0p, 0, hh);
    v16bf kf1 = load_fragB(k0p, 32, hh);
    v16bf kf2 = load_fragB(k1p, 0, hh);
    v16bf kf3 = load_fragB(k1p, 32, hh);
    v8f S0 = {}, S1 = {};
    S0 = __builtin_amdgcn_wmma_f32_16x16x32_bf16(false, a0, false, kf0, (short)0, S0, false, false);
    S0 = __builtin_amdgcn_wmma_f32_16x16x32_bf16(false, a1, false, kf1, (short)0, S0, false, false);
    S1 = __builtin_amdgcn_wmma_f32_16x16x32_bf16(false, a0, false, kf2, (short)0, S1, false, false);
    S1 = __builtin_amdgcn_wmma_f32_16x16x32_bf16(false, a1, false, kf3, (short)0, S1, false, false);
    // prefetch V fragments now; their dscnt wait hides behind the softmax VALU block
    v16bf vf[4];
#pragma unroll
    for (int t = 0; t < 4; ++t)
      vf[t] = load_fragB(&Vs[buf][(t * 16 + m) * 32], 0, hh);
#pragma unroll
    for (int r = 0; r < 8; ++r) {
      float s0 = S0[r] * 0.125f, s1 = S1[r] * 0.125f;   // 1/sqrt(64)
      float cm = fmaxf(s0, s1);
      cm = fmaxf(cm, __shfl_xor(cm, 1, 32));
      cm = fmaxf(cm, __shfl_xor(cm, 2, 32));
      cm = fmaxf(cm, __shfl_xor(cm, 4, 32));
      cm = fmaxf(cm, __shfl_xor(cm, 8, 32));
      float mn = fmaxf(mrow[r], cm);
      float sc = __expf(mrow[r] - mn);
      mrow[r] = mn;
      float p0 = __expf(s0 - mn), p1 = __expf(s1 - mn);
      float cs = p0 + p1;
      cs += __shfl_xor(cs, 1, 32);
      cs += __shfl_xor(cs, 2, 32);
      cs += __shfl_xor(cs, 4, 32);
      cs += __shfl_xor(cs, 8, 32);
      lrow[r] = lrow[r] * sc + cs;
#pragma unroll
      for (int t = 0; t < 4; ++t) O[t][r] *= sc;
      P[(r + 8 * hh) * 32 + m]      = (bf16)p0;   // per-wave private; DS ops in-order
      P[(r + 8 * hh) * 32 + 16 + m] = (bf16)p1;
    }
    v16bf pa = load_fragA(P + m * 32, 0, hh);
#pragma unroll
    for (int t = 0; t < 4; ++t)
      O[t] = __builtin_amdgcn_wmma_f32_16x16x32_bf16(false, pa, false, vf[t],
                                                     (short)0, O[t], false, false);
    wait_async0();
    __syncthreads();
    buf ^= 1;
  }
#pragma unroll
  for (int r = 0; r < 8; ++r) {
    float inv = 1.0f / lrow[r];
    size_t row = (size_t)b * NN + qbase + r + 8 * hh;
#pragma unroll
    for (int t = 0; t < 4; ++t)
      attn[row * DIM + h * HDIM + t * 16 + m] = (bf16)(O[t][r] * inv);
  }
}

extern "C" void kernel_launch(void* const* d_in, const int* in_sizes, int n_in,
                              void* d_out, int out_size, void* d_ws, size_t ws_size,
                              hipStream_t stream) {
  (void)in_sizes; (void)n_in; (void)out_size; (void)ws_size;
  const float* x     = (const float*)d_in[0];
  const float* ctx   = (const float*)d_in[1];
  const float* posm  = (const float*)d_in[2];
  const float* cposm = (const float*)d_in[3];
  const float* Wq    = (const float*)d_in[4];
  const float* Wkv   = (const float*)d_in[5];
  const float* Wout  = (const float*)d_in[6];
  const float* bout  = (const float*)d_in[7];
  float* out = (float*)d_out;

  char* w = (char*)d_ws;
  size_t off = 0;
  auto alloc = [&](size_t bytes) {
    char* p = w + off;
    off += (bytes + 255) & ~(size_t)255;
    return (void*)p;
  };
  bf16*  x_bf   = (bf16*) alloc((size_t)BB * NN * DIM * 2);
  bf16*  ctx_bf = (bf16*) alloc((size_t)BB * CTXN * CTXDIM * 2);
  bf16*  wq_t   = (bf16*) alloc((size_t)DIM * DIM * 2);
  bf16*  wkv_t  = (bf16*) alloc((size_t)CTXDIM * 2 * DIM * 2);
  bf16*  wout_t = (bf16*) alloc((size_t)DIM * DIM * 2);
  float* Qf     = (float*)alloc((size_t)BB * NN * DIM * 4);
  float* KVf    = (float*)alloc((size_t)BB * CTXN * 2 * DIM * 4);
  bf16*  Qb     = (bf16*) alloc((size_t)BB * NN * DIM * 2);
  bf16*  Kb     = (bf16*) alloc((size_t)BB * CTXN * DIM * 2);
  bf16*  Vt     = (bf16*) alloc((size_t)BB * CTXN * DIM * 2);
  bf16*  attn_b = (bf16*) alloc((size_t)BB * NN * DIM * 2);

  int n1 = BB * NN * DIM;
  cvt_bf16<<<(n1 + 255) / 256, 256, 0, stream>>>(x, x_bf, n1);
  int n2 = BB * CTXN * CTXDIM;
  cvt_bf16<<<(n2 + 255) / 256, 256, 0, stream>>>(ctx, ctx_bf, n2);
  cvt_tr<<<(DIM * DIM + 255) / 256, 256, 0, stream>>>(Wq, wq_t, DIM, DIM);
  cvt_tr<<<(CTXDIM * 2 * DIM + 255) / 256, 256, 0, stream>>>(Wkv, wkv_t, CTXDIM, 2 * DIM);
  cvt_tr<<<(DIM * DIM + 255) / 256, 256, 0, stream>>>(Wout, wout_t, DIM, DIM);

  gemm_bf16<<<dim3(DIM / 64, (BB * NN) / 128), 256, 0, stream>>>(
      x_bf, wq_t, Qf, nullptr, BB * NN, DIM, DIM);
  gemm_bf16<<<dim3((2 * DIM) / 64, (BB * CTXN) / 128), 256, 0, stream>>>(
      ctx_bf, wkv_t, KVf, nullptr, BB * CTXN, 2 * DIM, CTXDIM);

  int nr = BB * NN * NHEADS * 32;
  rope_q<<<(nr + 255) / 256, 256, 0, stream>>>(Qf, posm, Qb);
  rope_k<<<(nr + 255) / 256, 256, 0, stream>>>(KVf, cposm, Kb);
  int nv = BB * DIM * CTXN;
  vtrans<<<(nv + 255) / 256, 256, 0, stream>>>(KVf, Vt);

  attention<<<BB * NHEADS * 16, 256, 0, stream>>>(Qb, Kb, Vt, attn_b);

  gemm_bf16<<<dim3(DIM / 64, (BB * NN) / 128), 256, 0, stream>>>(
      attn_b, wout_t, out, bout, BB * NN, DIM, DIM);
}
